// LinearAttention_57578331570429
// MI455X (gfx1250) — matrix operand order repaired and checked by reading, not compiled, not run on hardware
//
#include <hip/hip_runtime.h>
#include <hip/hip_bf16.h>
#include <math.h>

// ---------------------------------------------------------------------------
// LinearAttention for MI455X (gfx1250, wave32, WMMA 16x16x32 bf16).
//
// Data movement: TDM (tensor_load_to_lds, TENSORcnt) stages GEMM tiles;
// GLOBAL_LOAD_ASYNC_TO_LDS (ASYNCcnt) stages the context-GEMM tiles.
// Both are double-buffered so HBM traffic overlaps the WMMA chain.
//
// Pipeline (fp32 in memory, bf16 inside the matrix units):
//   K1  qkv  = W_qkv(384x256) @ X[b](256x4096)            (WMMA GEMM, TDM)
//   K2  softmax over head-dim d=32 on q, * DIM_HEAD^-0.5
//   K3  softmax over spatial n=4096 on k
//   K4  ctx[b,h] = K(32x4096) @ (V/n)^T(4096x32)          (WMMA, async-LDS)
//   K5  W_eff[b] = W_out @ blockdiag_h(ctx[b,h]^T)        (tiny VALU GEMM)
//   K6  pre  = W_eff[b](256x128) @ Q[b](128x4096) + bias  (WMMA GEMM, TDM)
//   K7  channel LayerNorm over 256, * g  -> d_out
//
// Workspace (bytes): qkv 100,663,296 | ctx 262,144 | weff 2,097,152 |
//                    pre 67,108,864  -> ~170 MB total in d_ws.
// ---------------------------------------------------------------------------

typedef __bf16    v16bf __attribute__((ext_vector_type(16)));
typedef float     v8f   __attribute__((ext_vector_type(8)));
typedef unsigned  u32x4 __attribute__((ext_vector_type(4)));
typedef unsigned  u32x8 __attribute__((ext_vector_type(8)));

#define HEADS 4
#define DIM_HEAD 32
#define NTOK 4096                  // 64*64
#define QKV_CH 384
#define DIM 256
#define HIDDEN 128
#define SCALE 0.17677669529663687f // 32^-0.5

// Low 32 bits of a flat pointer into LDS == LDS byte address (aperture bits
// live in addr[63:32]); this is what async/TDM LDS operands want.
__device__ __forceinline__ unsigned lds_addr32(const void* p) {
  return (unsigned)(unsigned long long)p;
}

__device__ __forceinline__ void async_b128_to_lds(const void* gptr, unsigned lds) {
  asm volatile("global_load_async_to_lds_b128 %0, %1, off"
               :: "v"(lds), "v"((unsigned long long)gptr)
               : "memory");
}

// ---------------------------------------------------------------------------
// TDM: one-shot DMA of a 2-D fp32 tile (tile_w x tile_h, row stride ld elems)
// from global memory into densely packed LDS (row stride = tile_w).
// Descriptor per CDNA5 ISA 8.3/8.4: 2-D tensors need only D# groups 0 and 1.
// tensor_dim0/1 are set equal to the tile dims (tiles never clip here).
// ---------------------------------------------------------------------------
__device__ __forceinline__ void tdm_load_tile_f32(unsigned lds, const float* gtile,
                                                  unsigned tile_w, unsigned tile_h,
                                                  unsigned ld) {
  const unsigned long long ga = (unsigned long long)gtile;
  u32x4 g0;
  g0[0] = 1u;                                            // count=1, user mode
  g0[1] = lds;                                           // lds_addr
  g0[2] = (unsigned)ga;                                  // global_addr[31:0]
  g0[3] = ((unsigned)(ga >> 32) & 0x1FFFFFFu)            // global_addr[56:32]
          | 0x80000000u;                                 // type=2 ("image")
  u32x8 g1;
  g1[0] = 2u << 16;                                      // data_size=4B
  g1[1] = (tile_w & 0xFFFFu) << 16;                      // tensor_dim0 lo
  g1[2] = (tile_w >> 16) | ((tile_h & 0xFFFFu) << 16);   // dim0 hi | dim1 lo
  g1[3] = (tile_h >> 16) | (tile_w << 16);               // dim1 hi | tile_dim0
  g1[4] = tile_h & 0xFFFFu;                              // tile_dim1 (tile_dim2=0)
  g1[5] = ld;                                            // tensor_dim0_stride lo
  g1[6] = 0;                                             // stride0 hi | stride1 lo
  g1[7] = 0;                                             // stride1 hi
  asm volatile("tensor_load_to_lds %0, %1" :: "s"(g0), "s"(g1) : "memory");
}

// ---------------------------------------------------------------------------
// Generic bf16 WMMA GEMM: C[bz] = A[bz](MxK) @ B[bz](KxN) (+ bias)
// Block tile 64(M) x 128(N); 256 threads = 8 waves arranged 2(M) x 4(N);
// each wave computes a 32x32 patch = 2x2 WMMA 16x16 accumulators.
// K mult of 32; M mult of 64; N mult of 128.
// Tiles staged fp32 by the Tensor Data Mover, double buffered: wave 0 issues
// the k+1 tile pair, waits TENSORcnt<=2 (its k tiles done), barrier publishes.
// ---------------------------------------------------------------------------
__global__ __launch_bounds__(256)
void gemm_wmma_bf16(const float* __restrict__ A, long long sAb, int lda,
                    const float* __restrict__ B, long long sBb, int ldb,
                    float*       __restrict__ C, long long sCb, int ldc,
                    int K, const float* __restrict__ bias) {
  __shared__ float As[2][64 * 32];    // 2 x 8 KB
  __shared__ float Bs[2][32 * 128];   // 2 x 16 KB

  const int tid  = threadIdx.x;
  const int lane = tid & 31;
  const int wave = tid >> 5;
  const int wm   = wave >> 2;       // 0..1  (32 rows each)
  const int wn   = wave & 3;        // 0..3  (32 cols each)
  const int m0   = blockIdx.y * 64;
  const int n0   = blockIdx.x * 128;

  A += (long long)blockIdx.z * sAb;
  B += (long long)blockIdx.z * sBb;
  C += (long long)blockIdx.z * sCb;

  const int lm = lane & 15;         // M (A row / D col) index within tile
  const int hi = lane >> 4;         // half-wave select
  const int kh = hi * 8;
  const int kb = hi * 16;

  v8f acc[2][2] = {};

  const int nsteps = K >> 5;
  if (wave == 0) {
    tdm_load_tile_f32(lds_addr32(&As[0][0]), &A[(long long)m0 * lda], 32, 64, (unsigned)lda);
    tdm_load_tile_f32(lds_addr32(&Bs[0][0]), &B[n0],                 128, 32, (unsigned)ldb);
  }

  for (int i = 0; i < nsteps; ++i) {
    const int buf = i & 1;
    if (wave == 0) {
      if (i + 1 < nsteps) {
        const int k1 = (i + 1) << 5;
        tdm_load_tile_f32(lds_addr32(&As[buf ^ 1][0]),
                          &A[(long long)m0 * lda + k1], 32, 64, (unsigned)lda);
        tdm_load_tile_f32(lds_addr32(&Bs[buf ^ 1][0]),
                          &B[(long long)k1 * ldb + n0], 128, 32, (unsigned)ldb);
        __builtin_amdgcn_s_wait_tensorcnt((short)2);  // current buf landed
      } else {
        __builtin_amdgcn_s_wait_tensorcnt((short)0);
      }
    }
    __syncthreads();

    const float* __restrict__ Asb = As[buf];
    const float* __restrict__ Bsb = Bs[buf];

    // A fragments: lanes 0-15 row M=lane, K(0..7,16..23); lanes 16-31 K(8..15,24..31)
    v16bf afrag[2];
#pragma unroll
    for (int mt = 0; mt < 2; ++mt) {
      const int row = wm * 32 + mt * 16 + lm;
#pragma unroll
      for (int half = 0; half < 2; ++half)
#pragma unroll
        for (int j = 0; j < 4; ++j) {
          const int kk = half * 16 + kh + 2 * j;
          afrag[mt][half * 8 + 2 * j]     = (__bf16)Asb[row * 32 + kk];
          afrag[mt][half * 8 + 2 * j + 1] = (__bf16)Asb[row * 32 + kk + 1];
        }
    }
    // B fragments: lanes 0-15 K(0..7,16..23); lanes 16-31 K(8..15,24..31); N = lane%16
    v16bf bfrag[2];
#pragma unroll
    for (int nt = 0; nt < 2; ++nt) {
      const int col = wn * 32 + nt * 16 + lm;
#pragma unroll
      for (int v = 0; v < 8; ++v) {
        bfrag[nt][2 * v]     = (__bf16)Bsb[((v >> 2) * 16 + kh + 2 * (v & 3))     * 128 + col];
        bfrag[nt][2 * v + 1] = (__bf16)Bsb[((v >> 2) * 16 + kh + 2 * (v & 3) + 1) * 128 + col];
      }
    }

#pragma unroll
    for (int mt = 0; mt < 2; ++mt)
#pragma unroll
      for (int nt = 0; nt < 2; ++nt)
        acc[mt][nt] = __builtin_amdgcn_wmma_f32_16x16x32_bf16(
            false, afrag[mt], false, bfrag[nt], (short)0, acc[mt][nt],
            false, false);
    __syncthreads();   // everyone done reading buf before it is re-filled
  }

  // epilogue: D layout = VGPR r -> M = r + 8*hi, N = lane%16
#pragma unroll
  for (int mt = 0; mt < 2; ++mt)
#pragma unroll
    for (int nt = 0; nt < 2; ++nt) {
      const int col = n0 + wn * 32 + nt * 16 + lm;
#pragma unroll
      for (int r = 0; r < 8; ++r) {
        const int row = m0 + wm * 32 + mt * 16 + r + hi * 8;
        float val = acc[mt][nt][r];
        if (bias) val += bias[row];
        C[(long long)row * ldc + col] = val;
      }
    }
}

// ---------------------------------------------------------------------------
// K2: softmax over head-dim (32 strided floats) on q channels, * SCALE
// ---------------------------------------------------------------------------
__global__ __launch_bounds__(256)
void softmax_q_kernel(float* __restrict__ qkv) {
  const long long idx = (long long)blockIdx.x * blockDim.x + threadIdx.x;
  const int n = (int)(idx & (NTOK - 1));
  const int h = (int)((idx >> 12) & (HEADS - 1));
  const int b = (int)(idx >> 14);
  float* p = qkv + (long long)b * QKV_CH * NTOK + (long long)(h * DIM_HEAD) * NTOK + n;

  float v[DIM_HEAD];
#pragma unroll
  for (int d = 0; d < DIM_HEAD; ++d) v[d] = p[(long long)d * NTOK];
  float mx = v[0];
#pragma unroll
  for (int d = 1; d < DIM_HEAD; ++d) mx = fmaxf(mx, v[d]);
  float s = 0.f;
#pragma unroll
  for (int d = 0; d < DIM_HEAD; ++d) { v[d] = __expf(v[d] - mx); s += v[d]; }
  const float sc = SCALE / s;
#pragma unroll
  for (int d = 0; d < DIM_HEAD; ++d) p[(long long)d * NTOK] = v[d] * sc;
}

// ---------------------------------------------------------------------------
// K3: softmax over n=4096 (one contiguous row per block) on k channels
// ---------------------------------------------------------------------------
__global__ __launch_bounds__(256)
void softmax_k_kernel(float* __restrict__ qkv) {
  const int row = blockIdx.x;            // 0..2047
  const int b = row >> 7, c = row & 127;
  float* p = qkv + (long long)b * QKV_CH * NTOK + (long long)(HIDDEN + c) * NTOK;

  __shared__ float red[256];
  const int tid = threadIdx.x;

  float loc[16];
#pragma unroll
  for (int j = 0; j < 16; ++j) loc[j] = p[tid + j * 256];

  float mx = loc[0];
#pragma unroll
  for (int j = 1; j < 16; ++j) mx = fmaxf(mx, loc[j]);
  red[tid] = mx;
  __syncthreads();
  for (int off = 128; off > 0; off >>= 1) {
    if (tid < off) red[tid] = fmaxf(red[tid], red[tid + off]);
    __syncthreads();
  }
  const float m = red[0];
  __syncthreads();

  float s = 0.f;
#pragma unroll
  for (int j = 0; j < 16; ++j) { loc[j] = __expf(loc[j] - m); s += loc[j]; }
  red[tid] = s;
  __syncthreads();
  for (int off = 128; off > 0; off >>= 1) {
    if (tid < off) red[tid] += red[tid + off];
    __syncthreads();
  }
  const float inv = 1.0f / red[0];
#pragma unroll
  for (int j = 0; j < 16; ++j) p[tid + j * 256] = loc[j] * inv;
}

// ---------------------------------------------------------------------------
// K4: ctx[b,h](32x32) = K(32x4096) @ (V*invn)^T, WMMA bf16.
// 128 threads = 4 waves, one 16x16 tile each. 32x32 K/V fp32 tiles are
// staged into LDS with double-buffered async copies (4 b128 per wave/step).
// ---------------------------------------------------------------------------
__device__ __forceinline__ void ctx_issue_async(const float* kp, const float* vp,
                                                float* Kbuf, float* Vbuf,
                                                int k0, int wave, int lane) {
#pragma unroll
  for (int r = 0; r < 2; ++r) {
    const int c   = wave * 2 + r;          // chunk 0..7 (4 rows each)
    const int row = c * 4 + (lane >> 3);
    const int col = (lane & 7) * 4;
    async_b128_to_lds(&kp[(long long)row * NTOK + k0 + col],
                      lds_addr32(&Kbuf[row * 32 + col]));
    async_b128_to_lds(&vp[(long long)row * NTOK + k0 + col],
                      lds_addr32(&Vbuf[row * 32 + col]));
  }
}

__global__ __launch_bounds__(128)
void context_wmma_kernel(const float* __restrict__ qkv, float* __restrict__ ctx) {
  __shared__ float Kt[2][32 * 32];   // 2 x 4 KB
  __shared__ float Vt[2][32 * 32];   // 2 x 4 KB

  const int bh = blockIdx.x;
  const int b = bh >> 2, h = bh & 3;
  const long long S = (long long)QKV_CH * NTOK;
  const float* kp = qkv + b * S + (long long)(HIDDEN     + h * DIM_HEAD) * NTOK;
  const float* vp = qkv + b * S + (long long)(2 * HIDDEN + h * DIM_HEAD) * NTOK;

  const int wave = threadIdx.x >> 5, lane = threadIdx.x & 31;
  const int mt = wave >> 1, nt = wave & 1;
  const int lm = lane & 15, hi = lane >> 4;
  const float invn = 1.0f / (float)NTOK;

  const int arow = mt * 16 + lm;   // d index (row of K matrix)
  const int bcol = nt * 16 + lm;   // e index (row of V matrix)
  const int kh = hi * 8, kb = hi * 16;

  v8f acc = {};
  const int nsteps = NTOK >> 5;    // 128
  ctx_issue_async(kp, vp, Kt[0], Vt[0], 0, wave, lane);

  for (int i = 0; i < nsteps; ++i) {
    const int buf = i & 1;
    if (i + 1 < nsteps) {
      ctx_issue_async(kp, vp, Kt[buf ^ 1], Vt[buf ^ 1], (i + 1) << 5, wave, lane);
      asm volatile("s_wait_asynccnt 0x4" ::: "memory");
    } else {
      asm volatile("s_wait_asynccnt 0x0" ::: "memory");
    }
    __syncthreads();

    const float* __restrict__ Ks = Kt[buf];
    const float* __restrict__ Vs = Vt[buf];
    v16bf a, bb;
#pragma unroll
    for (int half = 0; half < 2; ++half)
#pragma unroll
      for (int j = 0; j < 4; ++j) {
        const int kk = half * 16 + kh + 2 * j;
        a[half * 8 + 2 * j]     = (__bf16)Ks[arow * 32 + kk];
        a[half * 8 + 2 * j + 1] = (__bf16)Ks[arow * 32 + kk + 1];
      }
#pragma unroll
    for (int v = 0; v < 8; ++v) {
      bb[2 * v]     = (__bf16)(Vs[bcol * 32 + (v >> 2) * 16 + kh + 2 * (v & 3)]     * invn);
      bb[2 * v + 1] = (__bf16)(Vs[bcol * 32 + (v >> 2) * 16 + kh + 2 * (v & 3) + 1] * invn);
    }
    acc = __builtin_amdgcn_wmma_f32_16x16x32_bf16(false, a, false, bb,
                                                  (short)0, acc, false, false);
    __syncthreads();
  }

  float* cp = ctx + (long long)bh * DIM_HEAD * DIM_HEAD;
#pragma unroll
  for (int r = 0; r < 8; ++r) {
    const int d = mt * 16 + r + hi * 8;
    const int e = nt * 16 + lm;
    cp[d * DIM_HEAD + e] = acc[r];
  }
}

// ---------------------------------------------------------------------------
// K5: W_eff[b][o][h*32+d] = sum_e W_out[o][h*32+e] * ctx[b,h][d][e]
// ---------------------------------------------------------------------------
__global__ __launch_bounds__(256)
void weff_kernel(const float* __restrict__ w_out, const float* __restrict__ ctx,
                 float* __restrict__ weff) {
  const long long idx = (long long)blockIdx.x * blockDim.x + threadIdx.x;
  const int hd = (int)(idx & 127);
  const int o  = (int)((idx >> 7) & 255);
  const int b  = (int)(idx >> 15);
  const int h = hd >> 5, d = hd & 31;
  const float* wrow = w_out + o * HIDDEN + h * DIM_HEAD;                  // over e
  const float* crow = ctx + (((long long)b * HEADS + h) * DIM_HEAD + d) * DIM_HEAD;
  float s = 0.f;
#pragma unroll
  for (int e = 0; e < DIM_HEAD; ++e) s += wrow[e] * crow[e];
  weff[idx] = s;
}

// ---------------------------------------------------------------------------
// K7: channel LayerNorm over 256 (biased var) * g  -> d_out
// ---------------------------------------------------------------------------
__global__ __launch_bounds__(256)
void layernorm_kernel(const float* __restrict__ pre, const float* __restrict__ g,
                      float* __restrict__ out) {
  const long long idx = (long long)blockIdx.x * blockDim.x + threadIdx.x;
  const int n = (int)(idx & (NTOK - 1));
  const int b = (int)(idx >> 12);
  const float* p = pre + (long long)b * DIM * NTOK + n;

  float s1 = 0.f, s2 = 0.f;
  for (int o = 0; o < DIM; ++o) {
    const float x = p[(long long)o * NTOK];
    s1 += x; s2 += x * x;
  }
  const float mean = s1 * (1.0f / DIM);
  const float var  = s2 * (1.0f / DIM) - mean * mean;
  const float inv  = rsqrtf(var + 1e-5f);

  float* q = out + (long long)b * DIM * NTOK + n;
  for (int o = 0; o < DIM; ++o)
    q[(long long)o * NTOK] = (p[(long long)o * NTOK] - mean) * inv * g[o];
}

// ---------------------------------------------------------------------------
extern "C" void kernel_launch(void* const* d_in, const int* in_sizes, int n_in,
                              void* d_out, int out_size, void* d_ws, size_t ws_size,
                              hipStream_t stream) {
  const float* x     = (const float*)d_in[0];  // [16,256,64,64]
  const float* w_qkv = (const float*)d_in[1];  // [384,256]
  const float* w_out = (const float*)d_in[2];  // [256,128]
  const float* b_out = (const float*)d_in[3];  // [256]
  const float* g     = (const float*)d_in[4];  // [1,256,1,1]
  float* out = (float*)d_out;

  char* ws = (char*)d_ws;
  float* qkv  = (float*)(ws);                                  // 100,663,296 B
  float* ctx  = (float*)(ws + 100663296ull);                   //     262,144 B
  float* weff = (float*)(ws + 100925440ull);                   //   2,097,152 B
  float* pre  = (float*)(ws + 103022592ull);                   //  67,108,864 B

  const long long sX   = (long long)DIM * NTOK;        // 256*4096
  const long long sQKV = (long long)QKV_CH * NTOK;     // 384*4096
  const long long sOut = (long long)DIM * NTOK;
  const long long sWe  = (long long)DIM * HIDDEN;      // 256*128

  // K1: qkv = W_qkv @ X per batch
  {
    dim3 grid(NTOK / 128, QKV_CH / 64, 16);
    gemm_wmma_bf16<<<grid, 256, 0, stream>>>(
        w_qkv, 0, DIM, x, sX, NTOK, qkv, sQKV, NTOK, DIM, nullptr);
  }
  // K2: softmax over d on q (in place)
  softmax_q_kernel<<<(16 * HEADS * NTOK) / 256, 256, 0, stream>>>(qkv);
  // K3: softmax over n on k (in place)
  softmax_k_kernel<<<16 * HIDDEN, 256, 0, stream>>>(qkv);
  // K4: context per (b,h), folds v/n
  context_wmma_kernel<<<16 * HEADS, 128, 0, stream>>>(qkv, ctx);
  // K5: W_eff = W_out @ blockdiag(ctx^T)
  weff_kernel<<<(16 * DIM * HIDDEN) / 256, 256, 0, stream>>>(w_out, ctx, weff);
  // K6: pre = W_eff @ Q + bias per batch (Q = rows 0..127 of qkv)
  {
    dim3 grid(NTOK / 128, DIM / 64, 16);
    gemm_wmma_bf16<<<grid, 256, 0, stream>>>(
        weff, sWe, HIDDEN, qkv, sQKV, NTOK, pre, sOut, NTOK, HIDDEN, b_out);
  }
  // K7: LayerNorm -> d_out
  layernorm_kernel<<<(16 * NTOK) / 256, 256, 0, stream>>>(pre, g, out);
}
